// MiniTransformer_73151882985759
// MI455X (gfx1250) — matrix operand order repaired and checked by reading, not compiled, not run on hardware
//
#include <hip/hip_runtime.h>
#include <hip/hip_bf16.h>
#include <cstdint>

typedef __bf16 bf16_t;
typedef __bf16 v16bf __attribute__((ext_vector_type(16)));
typedef __bf16 v8bf  __attribute__((ext_vector_type(8)));
typedef float  v8f   __attribute__((ext_vector_type(8)));
typedef unsigned int v4u __attribute__((ext_vector_type(4)));
typedef int v8i __attribute__((ext_vector_type(8)));
typedef int v4i __attribute__((ext_vector_type(4)));

constexpr int BB = 16, TT = 1024, DD = 512, HH = 8, LLAY = 6, VV = 96;
constexpr int HS = 64, FFD = 2048, BT = BB * TT;

__device__ __forceinline__ v8f wmma_bf16(v8f acc, v16bf a, v16bf b) {
    return __builtin_amdgcn_wmma_f32_16x16x32_bf16(
        false, a, false, b, (short)0, acc, false, false);
}

// A-fragment (16x32 bf16, row-major source): lane m=lane&15, half=lane>>4
// holds K = [half*16, half*16+16).
__device__ __forceinline__ v16bf load_a_frag(const bf16_t* row, int k0, int hf) {
    union { v16bf v; v8bf h[2]; } u;
    u.h[0] = *(const v8bf*)(row + k0 + hf * 16);
    u.h[1] = *(const v8bf*)(row + k0 + hf * 16 + 8);
    return u.v;
}

// ------------------------------------------------- TDM: 2D bf16 panel -> LDS
// Loads tile [tileN=128 rows x tileK=64 cols] of Bt[N,K] (row-major, 2-byte
// elems) starting at gaddr into LDS at lds_off, rows packed at 64-elem pitch.
__device__ __forceinline__ void tdm_load_panel(const bf16_t* gaddr, unsigned lds_off,
                                               int K, int N) {
#if __has_builtin(__builtin_amdgcn_tensor_load_to_lds)
    const unsigned long long ga = (unsigned long long)(uintptr_t)gaddr;
    v4u g0;
    g0[0] = 1u;                                    // count=1 (valid user D#)
    g0[1] = lds_off;                               // lds_addr (bytes)
    g0[2] = (unsigned)(ga & 0xFFFFFFFFu);          // global_addr[31:0]
    g0[3] = (unsigned)((ga >> 32) & 0x01FFFFFFu) | (2u << 30); // addr[56:32], type=2
    const unsigned td0 = (unsigned)K;              // tensor_dim0 (elems, contiguous)
    const unsigned td1 = (unsigned)N;              // tensor_dim1 (rows)
    const unsigned long long s0 = (unsigned long long)(unsigned)K; // dim0 stride
    v8i g1;
    g1[0] = (int)(1u << 16);                       // data_size=1 -> 2 bytes
    g1[1] = (int)((td0 & 0xFFFFu) << 16);          // tensor_dim0[15:0]
    g1[2] = (int)(((td0 >> 16) & 0xFFFFu) | ((td1 & 0xFFFFu) << 16));
    g1[3] = (int)(((td1 >> 16) & 0xFFFFu) | (64u << 16));  // tile_dim0 = 64
    g1[4] = (int)128;                              // tile_dim1=128, tile_dim2=0
    g1[5] = (int)(s0 & 0xFFFFFFFFu);
    g1[6] = (int)((s0 >> 32) & 0xFFFFu);           // stride0[47:32], stride1 lo=0
    g1[7] = 0;
    const v4i z4 = {0, 0, 0, 0};                   // groups 2/3 unused (2D)
    const v8i z8 = {0, 0, 0, 0, 0, 0, 0, 0};       // extra group (6-arg form)
    __builtin_amdgcn_tensor_load_to_lds(g0, g1, z4, z4, z8, 0);
#else
    (void)gaddr; (void)lds_off; (void)K; (void)N;
#endif
}

// ---------------------------------------------------------------- embedding
__global__ void embed_add(const int* __restrict__ idx, const float* __restrict__ tok,
                          const float* __restrict__ pos, float* __restrict__ x) {
    const int i = blockIdx.x * 256 + threadIdx.x;
    if (i >= BT * DD) return;
    const int row = i >> 9;
    const int c   = i & (DD - 1);
    const int t   = row & (TT - 1);
    x[i] = tok[(size_t)idx[row] * DD + c] + pos[(size_t)t * DD + c];
}

// ------------------------------------------------- fp32 -> bf16 transposed
// in: [G, K, Ng] fp32   out: [G, Ng, K] bf16
__global__ void conv_transpose_bf16(const float* __restrict__ in, bf16_t* __restrict__ out,
                                    int K, int Ng, long total) {
    const long i = (long)blockIdx.x * 256 + threadIdx.x;
    if (i >= total) return;
    const long kn  = (long)K * Ng;
    const int  g   = (int)(i / kn);
    const int  rem = (int)(i - (long)g * kn);
    const int  k   = rem / Ng;
    const int  n   = rem - k * Ng;
    out[((size_t)g * Ng + n) * K + k] = (bf16_t)in[i];
}

// ----------------------------------------------------------- layernorm(bf16)
__global__ __launch_bounds__(256) void layernorm_bf16(
    const float* __restrict__ x, const float* __restrict__ s,
    const float* __restrict__ b, bf16_t* __restrict__ out) {
    const int lane = threadIdx.x & 31;
    const int wave = threadIdx.x >> 5;
    const int row  = blockIdx.x * 8 + wave;
    const float* xr = x + (size_t)row * DD;
    float v[16];
    float sum = 0.0f;
#pragma unroll
    for (int i = 0; i < 16; ++i) { v[i] = xr[lane + i * 32]; sum += v[i]; }
#pragma unroll
    for (int d = 1; d < 32; d <<= 1) sum += __shfl_xor(sum, d, 32);
    const float mean = sum * (1.0f / DD);
    float var = 0.0f;
#pragma unroll
    for (int i = 0; i < 16; ++i) { float t = v[i] - mean; var += t * t; }
#pragma unroll
    for (int d = 1; d < 32; d <<= 1) var += __shfl_xor(var, d, 32);
    const float inv = rsqrtf(var * (1.0f / DD) + 1e-5f);
    bf16_t* orow = out + (size_t)row * DD;
#pragma unroll
    for (int i = 0; i < 16; ++i) {
        const int c = lane + i * 32;
        orow[c] = (bf16_t)((v[i] - mean) * inv * s[c] + b[c]);
    }
}

// -------------------------------------------------------- epilogue helper
__device__ __forceinline__ void gemm_store_tile(
    v8f acc, int row0, int col, int N, float bv, int relu,
    const float* __restrict__ resid, float* __restrict__ outF,
    bf16_t* __restrict__ outB, int bmode, int hf) {
#pragma unroll
    for (int r = 0; r < 8; ++r) {
        const int row = row0 + r + hf * 8;
        float v = acc[r] + bv;
        if (relu) v = fmaxf(v, 0.0f);
        if (resid) v += resid[(size_t)row * N + col];
        if (outF) outF[(size_t)row * N + col] = v;
        if (outB) {
            size_t oi;
            if (bmode == 0) {
                oi = (size_t)row * N + col;
            } else {
                const int b = row >> 10, t = row & (TT - 1);
                const int h = col >> 6,  c = col & (HS - 1);
                if (bmode == 1) oi = (((size_t)(b * HH + h)) * TT + t) * HS + c;
                else            oi = (((size_t)(b * HH + h)) * HS + c) * TT + t;
            }
            outB[oi] = (bf16_t)v;
        }
    }
}

// ------------------------ one 64-wide k-chunk of the 32x64 wave tile
__device__ __forceinline__ void gemm_compute_chunk(
    v8f (&acc)[2][4], const bf16_t* arow0, const bf16_t* arow1,
    const bf16_t* pan, int kc, int n0, int hf, int nn) {
#pragma unroll
    for (int kk = 0; kk < 64; kk += 32) {
        const v16bf a0 = load_a_frag(arow0, kc + kk, hf);
        const v16bf a1 = load_a_frag(arow1, kc + kk, hf);
#pragma unroll
        for (int j = 0; j < 4; ++j) {
            const v16bf bfr =
                *(const v16bf*)(pan + (n0 + j * 16 + nn) * 64 + kk + hf * 16);
            acc[0][j] = wmma_bf16(acc[0][j], a0, bfr);
            acc[1][j] = wmma_bf16(acc[1][j], a1, bfr);
        }
    }
}

// ------------------------------------------- big WMMA GEMM (TDM-staged B)
// Block tile 128(M) x 128(N); 8 waves = 4(M) x 2(N); wave tile 32x64.
// B panel [128 N-rows x 64 K] double-buffered in LDS, filled by the Tensor
// Data Mover overlapped with WMMA compute on the other buffer.
// Requires M%128==0, N%128==0, K%64==0.
__global__ __launch_bounds__(256) void gemm_bf16_wmma_big(
    const bf16_t* __restrict__ A, const bf16_t* __restrict__ Bt,
    const float* __restrict__ bias, const float* __restrict__ resid,
    float* __restrict__ outF, bf16_t* __restrict__ outB,
    int M, int N, int K, int relu, int bmode) {
    __shared__ __align__(128) bf16_t panel[2][128 * 64];   // 2 x 16 KB
    const int lane = threadIdx.x & 31;
    const int wave = threadIdx.x >> 5;
    const int hf   = lane >> 4;
    const int nn   = lane & 15;
    const int wm   = wave & 3;
    const int wn   = wave >> 2;
    const int blockN = blockIdx.x * 128;
    const int m0     = blockIdx.y * 128 + wm * 32;
    const int n0     = wn * 64;                            // within panel

    v8f acc[2][4];
#pragma unroll
    for (int i = 0; i < 2; ++i)
#pragma unroll
        for (int j = 0; j < 4; ++j) acc[i][j] = (v8f){};

    const bf16_t* arow0 = A + (size_t)(m0 + nn) * K;
    const bf16_t* arow1 = A + (size_t)(m0 + 16 + nn) * K;
    const bf16_t* bbase = Bt + (size_t)blockN * K;

#if __has_builtin(__builtin_amdgcn_tensor_load_to_lds)
    const unsigned ldsb0 = (unsigned)(uintptr_t)(&panel[0][0]);
    const unsigned ldsb1 = (unsigned)(uintptr_t)(&panel[1][0]);
    // prologue: fill buffer 0
    if (wave == 0) {
        tdm_load_panel(bbase, ldsb0, K, N);
        __builtin_amdgcn_s_wait_tensorcnt(0);
    }
    __syncthreads();
    int cur = 0;
    for (int kc = 0; kc < K; kc += 64) {
        const int nxt = cur ^ 1;
        if (wave == 0 && kc + 64 < K) {
            // async: DMA next panel while this one is consumed
            tdm_load_panel(bbase + kc + 64, cur ? ldsb0 : ldsb1, K, N);
        }
        gemm_compute_chunk(acc, arow0, arow1, &panel[cur][0], kc, n0, hf, nn);
        if (wave == 0) __builtin_amdgcn_s_wait_tensorcnt(0);
        __syncthreads();   // next buffer complete, current buffer free
        cur = nxt;
    }
#else
    for (int kc = 0; kc < K; kc += 64) {
        __syncthreads();
        for (int t = threadIdx.x; t < 128 * 8; t += 256) {   // 16B per copy
            const int rowp = t >> 3;
            const int cp   = (t & 7) * 8;
            *(v8bf*)(&panel[0][0] + rowp * 64 + cp) =
                *(const v8bf*)(bbase + (size_t)rowp * K + kc + cp);
        }
        __syncthreads();
        gemm_compute_chunk(acc, arow0, arow1, &panel[0][0], kc, n0, hf, nn);
    }
#endif

#pragma unroll
    for (int j = 0; j < 4; ++j) {
        const int col = blockN + n0 + j * 16 + nn;
        const float bv = bias ? bias[col] : 0.0f;
        gemm_store_tile(acc[0][j], m0,      col, N, bv, relu, resid, outF, outB, bmode, hf);
        gemm_store_tile(acc[1][j], m0 + 16, col, N, bv, relu, resid, outF, outB, bmode, hf);
    }
}

// ------------------------------------ small WMMA GEMM (any N%16, used for logits)
__global__ __launch_bounds__(256) void gemm_bf16_wmma_small(
    const bf16_t* __restrict__ A, const bf16_t* __restrict__ Bt,
    const float* __restrict__ bias, float* __restrict__ outF,
    int M, int N, int K) {
    const int lane = threadIdx.x & 31;
    const int wave = threadIdx.x >> 5;
    const int hf   = lane >> 4;
    const int nn   = lane & 15;
    const int n0   = blockIdx.x * 16;
    const int m0   = blockIdx.y * 128 + wave * 16;

    v8f acc = {};
    const bf16_t* arow = A + (size_t)(m0 + nn) * K;
    const bf16_t* brow = Bt + (size_t)(n0 + nn) * K + hf * 16;
    for (int k0 = 0; k0 < K; k0 += 32) {
        acc = wmma_bf16(acc, load_a_frag(arow, k0, hf), *(const v16bf*)(brow + k0));
    }
    const float bv  = bias ? bias[n0 + nn] : 0.0f;
    const int   col = n0 + nn;
#pragma unroll
    for (int r = 0; r < 8; ++r) {
        outF[(size_t)(m0 + r + hf * 8) * N + col] = acc[r] + bv;
    }
}

// --------------------------------------------------- flash attention (WMMA)
// q,k: [B,H,T,HS] bf16 ; vt: [B,H,HS,T] bf16 ; o: [B,T,D] bf16 (head-concat)
__global__ __launch_bounds__(256) void attn_flash_wmma(
    const bf16_t* __restrict__ q, const bf16_t* __restrict__ kmat,
    const bf16_t* __restrict__ vt, bf16_t* __restrict__ o) {
    __shared__ __align__(64) bf16_t pbuf[8][16 * 32];
    const int lane = threadIdx.x & 31;
    const int wave = threadIdx.x >> 5;
    const int hf   = lane >> 4;
    const int nn   = lane & 15;
    const int bh   = blockIdx.x;
    const int r0   = blockIdx.y * 128 + wave * 16;
    const float scale = 0.044194173824159216f;  // D^-0.5 (full channel dim)

    const bf16_t* qrow = q + ((size_t)bh * TT + r0 + nn) * HS;
    const v16bf A0 = load_a_frag(qrow, 0, hf);
    const v16bf A1 = load_a_frag(qrow, 32, hf);

    v8f O0 = {}, O1 = {}, O2 = {}, O3 = {};
    float mrun[8], lrun[8];
#pragma unroll
    for (int r = 0; r < 8; ++r) { mrun[r] = -1e30f; lrun[r] = 0.0f; }

    const int send = r0 + 16;
    for (int sc = 0; sc < send; sc += 32) {
        const bf16_t* kl = kmat + ((size_t)bh * TT + sc + nn) * HS + hf * 16;
        const bf16_t* kr = kl + 16 * HS;
        v8f SL = {}, SR = {};
        SL = wmma_bf16(SL, A0, *(const v16bf*)(kl));
        SL = wmma_bf16(SL, A1, *(const v16bf*)(kl + 32));
        SR = wmma_bf16(SR, A0, *(const v16bf*)(kr));
        SR = wmma_bf16(SR, A1, *(const v16bf*)(kr + 32));

        float alpha[8];
#pragma unroll
        for (int r = 0; r < 8; ++r) {
            const int row = r0 + r + hf * 8;
            float sl = (sc + nn      <= row) ? SL[r] * scale : -1e30f;
            float sr = (sc + 16 + nn <= row) ? SR[r] * scale : -1e30f;
            float mx = fmaxf(sl, sr);
#pragma unroll
            for (int d = 1; d < 16; d <<= 1) mx = fmaxf(mx, __shfl_xor(mx, d, 32));
            const float mnew = fmaxf(mrun[r], mx);
            alpha[r] = __expf(mrun[r] - mnew);
            mrun[r]  = mnew;
            const float pl = __expf(sl - mnew);
            const float pr = __expf(sr - mnew);
            float rs = pl + pr;
#pragma unroll
            for (int d = 1; d < 16; d <<= 1) rs += __shfl_xor(rs, d, 32);
            lrun[r] = lrun[r] * alpha[r] + rs;
            pbuf[wave][(r + hf * 8) * 32 + nn]      = (bf16_t)pl;
            pbuf[wave][(r + hf * 8) * 32 + 16 + nn] = (bf16_t)pr;
        }
        // wave-internal LDS RAW: explicit CDNA5 dscnt wait before re-reading P
        asm volatile("s_wait_dscnt 0x0" ::: "memory");
        const v16bf Ap = load_a_frag(&pbuf[wave][nn * 32], 0, hf);
#pragma unroll
        for (int r = 0; r < 8; ++r) {
            O0[r] *= alpha[r]; O1[r] *= alpha[r];
            O2[r] *= alpha[r]; O3[r] *= alpha[r];
        }
        const bf16_t* vb = vt + ((size_t)bh * HS + nn) * TT + sc + hf * 16;
        O0 = wmma_bf16(O0, Ap, *(const v16bf*)(vb));
        O1 = wmma_bf16(O1, Ap, *(const v16bf*)(vb + 16 * TT));
        O2 = wmma_bf16(O2, Ap, *(const v16bf*)(vb + 32 * TT));
        O3 = wmma_bf16(O3, Ap, *(const v16bf*)(vb + 48 * TT));
    }
    const int b = bh >> 3, h = bh & 7;
#pragma unroll
    for (int r = 0; r < 8; ++r) {
        const float inv = 1.0f / lrun[r];
        const int t = r0 + r + hf * 8;
        bf16_t* op = o + ((size_t)(b * TT + t)) * DD + h * HS;
        op[nn]      = (bf16_t)(O0[r] * inv);
        op[16 + nn] = (bf16_t)(O1[r] * inv);
        op[32 + nn] = (bf16_t)(O2[r] * inv);
        op[48 + nn] = (bf16_t)(O3[r] * inv);
    }
}

// ---------------------------------------------------------------- NLL loss
__global__ void zero_one(float* p) { if (threadIdx.x == 0 && blockIdx.x == 0) *p = 0.0f; }

__global__ __launch_bounds__(256) void loss_nll(const float* __restrict__ logits,
                                                const int* __restrict__ tgt,
                                                float* __restrict__ loss) {
    const int lane = threadIdx.x & 31;
    const int wave = threadIdx.x >> 5;
    const int row  = blockIdx.x * 8 + wave;
    const float* lr = logits + (size_t)row * VV;
    const float v0 = lr[lane], v1 = lr[lane + 32], v2 = lr[lane + 64];
    float mx = fmaxf(v0, fmaxf(v1, v2));
#pragma unroll
    for (int d = 1; d < 32; d <<= 1) mx = fmaxf(mx, __shfl_xor(mx, d, 32));
    float se = __expf(v0 - mx) + __expf(v1 - mx) + __expf(v2 - mx);
#pragma unroll
    for (int d = 1; d < 32; d <<= 1) se += __shfl_xor(se, d, 32);
    if (lane == 0) {
        const float lse = mx + __logf(se);
        atomicAdd(loss, (lse - lr[tgt[row]]) * (1.0f / BT));
    }
}

// ------------------------------------------------------------------- host
extern "C" void kernel_launch(void* const* d_in, const int* in_sizes, int n_in,
                              void* d_out, int out_size, void* d_ws, size_t ws_size,
                              hipStream_t stream) {
    (void)in_sizes; (void)n_in; (void)out_size; (void)ws_size;
    const int*   idx   = (const int*)  d_in[0];
    const int*   tgt   = (const int*)  d_in[1];
    const float* tok   = (const float*)d_in[2];
    const float* pos   = (const float*)d_in[3];
    const float* Wq    = (const float*)d_in[4];
    const float* Wk    = (const float*)d_in[5];
    const float* Wv    = (const float*)d_in[6];
    const float* Wproj = (const float*)d_in[7];
    const float* bproj = (const float*)d_in[8];
    const float* W1    = (const float*)d_in[9];
    const float* b1    = (const float*)d_in[10];
    const float* W2    = (const float*)d_in[11];
    const float* b2    = (const float*)d_in[12];
    const float* ln1s  = (const float*)d_in[13];
    const float* ln1b  = (const float*)d_in[14];
    const float* ln2s  = (const float*)d_in[15];
    const float* ln2b  = (const float*)d_in[16];
    const float* lnfs  = (const float*)d_in[17];
    const float* lnfb  = (const float*)d_in[18];
    const float* Wout  = (const float*)d_in[19];
    const float* bout  = (const float*)d_in[20];

    char* ws = (char*)d_ws;
    size_t off = 0;
    auto carve = [&](size_t bytes) -> void* {
        void* p = ws + off;
        off += (bytes + 255) & ~(size_t)255;
        return p;
    };
    float*  xbuf  = (float*) carve((size_t)BT * DD * 4);
    bf16_t* hbf   = (bf16_t*)carve((size_t)BT * DD * 2);
    bf16_t* qb    = (bf16_t*)carve((size_t)BT * DD * 2);
    bf16_t* kb    = (bf16_t*)carve((size_t)BT * DD * 2);
    bf16_t* vtb   = (bf16_t*)carve((size_t)BT * DD * 2);
    bf16_t* obf   = (bf16_t*)carve((size_t)BT * DD * 2);
    bf16_t* ffb   = (bf16_t*)carve((size_t)BT * FFD * 2);
    bf16_t* WqB   = (bf16_t*)carve((size_t)LLAY * DD * DD * 2);
    bf16_t* WkB   = (bf16_t*)carve((size_t)LLAY * DD * DD * 2);
    bf16_t* WvB   = (bf16_t*)carve((size_t)LLAY * DD * DD * 2);
    bf16_t* WpB   = (bf16_t*)carve((size_t)LLAY * DD * DD * 2);
    bf16_t* W1B   = (bf16_t*)carve((size_t)LLAY * DD * FFD * 2);
    bf16_t* W2B   = (bf16_t*)carve((size_t)LLAY * DD * FFD * 2);
    bf16_t* WoutB = (bf16_t*)carve((size_t)DD * VV * 2);

    const dim3 blk(256);
    const long nqkv = (long)LLAY * HH * DD * HS;
    conv_transpose_bf16<<<(unsigned)((nqkv + 255) / 256), blk, 0, stream>>>(Wq, WqB, DD, HS, nqkv);
    conv_transpose_bf16<<<(unsigned)((nqkv + 255) / 256), blk, 0, stream>>>(Wk, WkB, DD, HS, nqkv);
    conv_transpose_bf16<<<(unsigned)((nqkv + 255) / 256), blk, 0, stream>>>(Wv, WvB, DD, HS, nqkv);
    const long nproj = (long)LLAY * DD * DD;
    conv_transpose_bf16<<<(unsigned)((nproj + 255) / 256), blk, 0, stream>>>(Wproj, WpB, DD, DD, nproj);
    const long nff = (long)LLAY * DD * FFD;
    conv_transpose_bf16<<<(unsigned)((nff + 255) / 256), blk, 0, stream>>>(W1, W1B, DD, FFD, nff);
    conv_transpose_bf16<<<(unsigned)((nff + 255) / 256), blk, 0, stream>>>(W2, W2B, FFD, DD, nff);
    const long nout = (long)DD * VV;
    conv_transpose_bf16<<<(unsigned)((nout + 255) / 256), blk, 0, stream>>>(Wout, WoutB, DD, VV, nout);

    embed_add<<<(BT * DD) / 256, blk, 0, stream>>>(idx, tok, pos, xbuf);

    auto gbig = [](int M, int N) { return dim3((unsigned)(N / 128), (unsigned)(M / 128)); };

    for (int l = 0; l < LLAY; ++l) {
        layernorm_bf16<<<BT / 8, blk, 0, stream>>>(xbuf, ln1s + l * DD, ln1b + l * DD, hbf);
        gemm_bf16_wmma_big<<<gbig(BT, DD), blk, 0, stream>>>(
            hbf, WqB + (size_t)l * DD * DD, nullptr, nullptr, nullptr, qb, BT, DD, DD, 0, 1);
        gemm_bf16_wmma_big<<<gbig(BT, DD), blk, 0, stream>>>(
            hbf, WkB + (size_t)l * DD * DD, nullptr, nullptr, nullptr, kb, BT, DD, DD, 0, 1);
        gemm_bf16_wmma_big<<<gbig(BT, DD), blk, 0, stream>>>(
            hbf, WvB + (size_t)l * DD * DD, nullptr, nullptr, nullptr, vtb, BT, DD, DD, 0, 2);
        attn_flash_wmma<<<dim3(BB * HH, TT / 128), blk, 0, stream>>>(qb, kb, vtb, obf);
        gemm_bf16_wmma_big<<<gbig(BT, DD), blk, 0, stream>>>(
            obf, WpB + (size_t)l * DD * DD, bproj + l * DD, xbuf, xbuf, nullptr, BT, DD, DD, 0, 0);
        layernorm_bf16<<<BT / 8, blk, 0, stream>>>(xbuf, ln2s + l * DD, ln2b + l * DD, hbf);
        gemm_bf16_wmma_big<<<gbig(BT, FFD), blk, 0, stream>>>(
            hbf, W1B + (size_t)l * DD * FFD, b1 + l * FFD, nullptr, nullptr, ffb, BT, FFD, DD, 1, 0);
        gemm_bf16_wmma_big<<<gbig(BT, DD), blk, 0, stream>>>(
            ffb, W2B + (size_t)l * DD * FFD, b2 + l * DD, xbuf, xbuf, nullptr, BT, DD, FFD, 0, 0);
    }

    layernorm_bf16<<<BT / 8, blk, 0, stream>>>(xbuf, lnfs, lnfb, hbf);

    float* logits = (float*)d_out;
    gemm_bf16_wmma_small<<<dim3(VV / 16, BT / 128), blk, 0, stream>>>(
        hbf, WoutB, bout, logits, BT, VV, DD);

    float* lossp = logits + (size_t)BT * VV;
    zero_one<<<1, 1, 0, stream>>>(lossp);
    loss_nll<<<BT / 8, blk, 0, stream>>>(logits, tgt, lossp);
}
